// CrystalFlowModel_67190468378981
// MI455X (gfx1250) — matrix operand order repaired and checked by reading, not compiled, not run on hardware
//
#include <hip/hip_runtime.h>
#include <hip/hip_bf16.h>

// Problem sizes (fixed by the reference).
#define NATOMS 20000
#define NEDGE  400000
#define BATCHB 32
#define HD     128
#define NL     4
#define NGAUSS 50
#define NGPAD  64
#define NTYPE  100
#define CUTV   5.0f

// f16 weight-region offsets (in halfs) inside the w16 workspace blob.
// Originals are row-major [K][N] (used by backward, where they act as [N][K]);
// *T copies are [N][K] (used by forward) so every fragment load is contiguous.
#define O_E1   0                        // eW1 padded   L x [64][128]
#define O_E2   32768                    // eW2          L x [128][128]
#define O_N1   98304                    // nW1          L x [256][128]
#define O_N2   229376                   // nW2          L x [128][128]
#define O_CWT  294912                   // comb_W^T     [128][256]
#define O_E1T  327680                   // eW1^T        L x [128][64]
#define O_E2T  360448                   // eW2^T        L x [128][128]
#define O_N1T  425984                   // nW1^T        L x [128][256]
#define O_N2T  557056                   // nW2^T        L x [128][128]
#define W16_TOT 622592

typedef __attribute__((ext_vector_type(16))) _Float16 v16h;
typedef __attribute__((ext_vector_type(8)))  float    v8f;

__device__ __forceinline__ float silu_f(float x){ float s = 1.0f/(1.0f+__expf(-x)); return x*s; }
__device__ __forceinline__ float dsilu_f(float x){ float s = 1.0f/(1.0f+__expf(-x)); return s*(1.0f + x*(1.0f-s)); }

// ---- Async global->LDS staging (CDNA5 ASYNCcnt path) -----------------------
__device__ __forceinline__ void async_g2l(_Float16* ldsdst, const _Float16* gsrc, int halfs){
  for (int o = (int)threadIdx.x*8; o < halfs; o += 256*8){
    unsigned loff = (unsigned)(uintptr_t)(ldsdst + o);   // low 32b of generic = LDS offset
    const _Float16* gp = gsrc + o;
    asm volatile("global_load_async_to_lds_b128 %0, %1, off"
                 :: "v"(loff), "v"(gp) : "memory");
  }
}
__device__ __forceinline__ void async_wait_barrier(){
  asm volatile("s_wait_asynccnt 0x0" ::: "memory");
  __syncthreads();
}

// ---- WMMA fragment loaders (wave32, 16x16x32 f16 -> f32) -------------------
// A tile: 16 rows x 32 K from row-major f16 LDS buffer with row stride lda.
__device__ __forceinline__ v16h ldA(const _Float16* A, int lda, int kbase){
  int lane = threadIdx.x & 31;
  int row  = lane & 15;
  int k0   = kbase + ((lane & 16) ? 8 : 0);
  const _Float16* p = A + (size_t)row*lda + k0;
  v16h a;
#pragma unroll
  for (int i=0;i<8;i++){ a[i]=p[i]; a[8+i]=p[i+16]; }
  return a;
}
// B tile from [N][K]-layout weights: per-lane 2x16B contiguous (b128-friendly).
__device__ __forceinline__ v16h ldBc(const _Float16* W, int ldk, int kbase, int nbase){
  int lane = threadIdx.x & 31;
  int n  = nbase + (lane & 15);
  int k0 = kbase + ((lane & 16) ? 8 : 0);
  const _Float16* p = W + (size_t)n*ldk + k0;
  v16h b;
#pragma unroll
  for (int i=0;i<8;i++){ b[i]=p[i]; b[8+i]=p[i+16]; }
  return b;
}
__device__ __forceinline__ v8f wmma32(v16h a, v16h b, v8f c){
  return __builtin_amdgcn_wmma_f32_16x16x32_f16(false, a, false, b, (short)0, c, false, false);
}
// C/D layout: col = nbase + (lane&15); rows = ((lane&16)?8:0)+i for c[i].

// ---------------------------------------------------------------------------
__global__ void k_zero(float* __restrict__ p, int n){
  int i = blockIdx.x*blockDim.x + threadIdx.x;
  if (i < n) p[i] = 0.0f;
}

__global__ void k_prep_w16(const float* __restrict__ combW, const float* __restrict__ eW1,
                           const float* __restrict__ eW2, const float* __restrict__ nW1,
                           const float* __restrict__ nW2, _Float16* __restrict__ w16){
  int i = blockIdx.x*blockDim.x + threadIdx.x;
  if (i < NL*NGPAD*HD){                                     // O_E1: [l][g][j], K-pad 50->64
    int l = i/(NGPAD*HD), r = i%(NGPAD*HD), g = r/HD, j = r%HD;
    float v = (g < NGAUSS) ? eW1[((size_t)l*NGAUSS + g)*HD + j] : 0.0f;
    w16[O_E1 + i] = (_Float16)v; return;
  }
  i -= NL*NGPAD*HD;
  if (i < NL*HD*HD){ w16[O_E2 + i] = (_Float16)eW2[i]; return; }
  i -= NL*HD*HD;
  if (i < NL*2*HD*HD){ w16[O_N1 + i] = (_Float16)nW1[i]; return; }
  i -= NL*2*HD*HD;
  if (i < NL*HD*HD){ w16[O_N2 + i] = (_Float16)nW2[i]; return; }
  i -= NL*HD*HD;
  if (i < 256*HD){                                          // O_CWT: [o][c]
    int o = i/256, c = i%256;
    w16[O_CWT + i] = (_Float16)combW[(size_t)c*HD + o]; return;
  }
  i -= 256*HD;
  if (i < NL*HD*NGPAD){                                     // O_E1T: [l][j][g]
    int l = i/(HD*NGPAD), r = i%(HD*NGPAD), j = r/NGPAD, g = r%NGPAD;
    float v = (g < NGAUSS) ? eW1[((size_t)l*NGAUSS + g)*HD + j] : 0.0f;
    w16[O_E1T + i] = (_Float16)v; return;
  }
  i -= NL*HD*NGPAD;
  if (i < NL*HD*HD){                                        // O_E2T: [l][j][i2]
    int l = i/(HD*HD), r = i%(HD*HD), j = r/HD, i2 = r%HD;
    w16[O_E2T + i] = (_Float16)eW2[((size_t)l*HD + i2)*HD + j]; return;
  }
  i -= NL*HD*HD;
  if (i < NL*HD*2*HD){                                      // O_N1T: [l][j][c]
    int l = i/(HD*2*HD), r = i%(HD*2*HD), j = r/(2*HD), c = r%(2*HD);
    w16[O_N1T + i] = (_Float16)nW1[((size_t)l*2*HD + c)*HD + j]; return;
  }
  i -= NL*HD*2*HD;
  if (i < NL*HD*HD){                                        // O_N2T: [l][j][i2]
    int l = i/(HD*HD), r = i%(HD*HD), j = r/HD, i2 = r%HD;
    w16[O_N2T + i] = (_Float16)nW2[((size_t)l*HD + i2)*HD + j];
  }
}

// Edge geometry: minimum-image vector, distance, padded f16 Gaussian features.
__global__ void k_edge_geom(const float* __restrict__ frac, const float* __restrict__ lattice,
                            const int* __restrict__ ei, const int* __restrict__ batch,
                            float* __restrict__ evec, float* __restrict__ edist,
                            _Float16* __restrict__ gauss){
  int e = blockIdx.x*blockDim.x + threadIdx.x;
  if (e >= NEDGE) return;
  int s = ei[e], d = ei[NEDGE + e];
  int b = batch[s];
  float dx[3];
#pragma unroll
  for (int j=0;j<3;j++){
    float t = frac[s*3+j] - frac[d*3+j] + 0.5f;
    dx[j] = t - floorf(t) - 0.5f;
  }
  const float* lat = lattice + b*9;
  float ev[3];
#pragma unroll
  for (int k=0;k<3;k++){
    ev[k] = dx[0]*lat[0*3+k] + dx[1]*lat[1*3+k] + dx[2]*lat[2*3+k];
    evec[e*3+k] = ev[k];
  }
  float dist = sqrtf(ev[0]*ev[0] + ev[1]*ev[1] + ev[2]*ev[2]);
  edist[e] = dist;
  const float step  = CUTV/(NGAUSS-1);
  const float coeff = -0.5f/(step*step);
  for (int g=0; g<NGPAD; g++){
    float v = 0.0f;
    if (g < NGAUSS){ float t = dist - step*g; v = __expf(coeff*t*t); }
    gauss[(size_t)e*NGPAD + g] = (_Float16)v;
  }
}

// h0 = [atom_emb[x], frac@coordW+b] @ comb_W + comb_b   (WMMA, 16 nodes/block)
__global__ void k_h0(const int* __restrict__ xt, const float* __restrict__ frac,
                     const float* __restrict__ emb, const float* __restrict__ coordW,
                     const float* __restrict__ coordb, const _Float16* __restrict__ w16,
                     const float* __restrict__ combb, float* __restrict__ h0){
  __shared__ __align__(16) _Float16 A[16][256];
  int base = blockIdx.x*16, tid = threadIdx.x;
  for (int idx = tid; idx < 16*256; idx += 256){
    int r = idx >> 8, c = idx & 255;
    int n = base + r;
    float v;
    if (c < HD) v = emb[(size_t)xt[n]*HD + c];
    else {
      int j = c - HD;
      v = frac[n*3+0]*coordW[0*HD+j] + frac[n*3+1]*coordW[1*HD+j] +
          frac[n*3+2]*coordW[2*HD+j] + coordb[j];
    }
    A[r][c] = (_Float16)v;
  }
  __syncthreads();
  int wave = tid >> 5, lane = tid & 31, nb = wave*16;
  v8f c = {};
  for (int k=0;k<256;k+=32) c = wmma32(ldA(&A[0][0], 256, k), ldBc(w16 + O_CWT, 256, k, nb), c);
  int col = nb + (lane & 15), rb = (lane & 16) ? 8 : 0;
#pragma unroll
  for (int i=0;i<8;i++) h0[(size_t)(base+rb+i)*HD + col] = c[i] + combb[col];
}

// Edge MLP forward fused with message scatter:
//   w = silu(g@eW1+b1)@eW2+b2 ; atomicAdd(agg[dst], h[src]*w)
// Gaussian tile + transposed weights staged via async global->LDS copies.
__global__ void k_edge_fwd(int l, const _Float16* __restrict__ gauss,
                           const _Float16* __restrict__ w16,
                           const float* __restrict__ eb1, const float* __restrict__ eb2,
                           const int* __restrict__ ei, const float* __restrict__ h,
                           float* __restrict__ agg){
  __shared__ __align__(16) _Float16 Gt[16][NGPAD];
  __shared__ __align__(16) _Float16 S[16][HD];
  __shared__ __align__(16) _Float16 W1s[HD*NGPAD];   // eW1^T  [j][g]
  __shared__ __align__(16) _Float16 W2s[HD*HD];      // eW2^T  [j][i]
  __shared__ int srcS[16], dstS[16];
  int base = blockIdx.x*16, tid = threadIdx.x;
  async_g2l(&Gt[0][0], gauss + (size_t)base*NGPAD, 16*NGPAD);
  async_g2l(W1s, w16 + O_E1T + (size_t)l*HD*NGPAD, HD*NGPAD);
  async_g2l(W2s, w16 + O_E2T + (size_t)l*HD*HD, HD*HD);
  if (tid < 16){ srcS[tid] = ei[base+tid]; dstS[tid] = ei[NEDGE+base+tid]; }
  if (tid == 0 && base + 16 < NEDGE)
    __builtin_prefetch(gauss + (size_t)(base+16)*NGPAD, 0, 1);   // global_prefetch
  async_wait_barrier();
  int wave = tid >> 5, lane = tid & 31, nb = wave*16;
  v8f c = {};
  for (int k=0;k<NGPAD;k+=32) c = wmma32(ldA(&Gt[0][0], NGPAD, k), ldBc(W1s, NGPAD, k, nb), c);
  int col = nb + (lane & 15), rb = (lane & 16) ? 8 : 0;
#pragma unroll
  for (int i=0;i<8;i++){
    float z = c[i] + eb1[l*HD + col];
    S[rb+i][col] = (_Float16)silu_f(z);
  }
  __syncthreads();
  v8f c2 = {};
  for (int k=0;k<HD;k+=32) c2 = wmma32(ldA(&S[0][0], HD, k), ldBc(W2s, HD, k, nb), c2);
#pragma unroll
  for (int i=0;i<8;i++){
    int r = rb+i;
    float we = c2[i] + eb2[l*HD + col];
    float m  = h[(size_t)srcS[r]*HD + col] * we;
    atomicAdd(&agg[(size_t)dstS[r]*HD + col], m);
  }
}

// Node MLP forward + residual + LayerNorm; stores u (pre-silu), y, mean/rstd.
__global__ void k_node_fwd(int l, const float* __restrict__ h, const float* __restrict__ agg,
                           const _Float16* __restrict__ w16,
                           const float* __restrict__ nb1, const float* __restrict__ nb2,
                           const float* __restrict__ lng, const float* __restrict__ lnb,
                           float* __restrict__ ubuf, float* __restrict__ ybuf,
                           float* __restrict__ mrbuf, float* __restrict__ hnext){
  __shared__ __align__(16) _Float16 A[16][256];
  __shared__ __align__(16) _Float16 S[16][HD];
  __shared__ float Y[16][HD];
  __shared__ float rowm[16], rowr[16];
  int base = blockIdx.x*16, tid = threadIdx.x;
  for (int idx = tid; idx < 16*256; idx += 256){
    int r = idx >> 8, c = idx & 255;
    float v = (c < HD) ? h[(size_t)(base+r)*HD + c] : agg[(size_t)(base+r)*HD + (c-HD)];
    A[r][c] = (_Float16)v;
  }
  __syncthreads();
  int wave = tid >> 5, lane = tid & 31, nb = wave*16;
  const _Float16* W1T = w16 + O_N1T + (size_t)l*HD*2*HD;   // [j][c] ldk=256
  const _Float16* W2T = w16 + O_N2T + (size_t)l*HD*HD;     // [j][i] ldk=128
  v8f c = {};
  for (int k=0;k<256;k+=32) c = wmma32(ldA(&A[0][0], 256, k), ldBc(W1T, 256, k, nb), c);
  int col = nb + (lane & 15), rb = (lane & 16) ? 8 : 0;
#pragma unroll
  for (int i=0;i<8;i++){
    float u = c[i] + nb1[l*HD + col];
    ubuf[(size_t)(base+rb+i)*HD + col] = u;
    S[rb+i][col] = (_Float16)silu_f(u);
  }
  __syncthreads();
  v8f c2 = {};
  for (int k=0;k<HD;k+=32) c2 = wmma32(ldA(&S[0][0], HD, k), ldBc(W2T, HD, k, nb), c2);
  float yv[8];
#pragma unroll
  for (int i=0;i<8;i++){
    int r = rb+i, n = base+r;
    float y = h[(size_t)n*HD + col] + c2[i] + nb2[l*HD + col];
    yv[i] = y; Y[r][col] = y; ybuf[(size_t)n*HD + col] = y;
  }
  __syncthreads();
  if (tid < 16){
    float s = 0.f, s2 = 0.f;
    for (int k2=0;k2<HD;k2++){ float y = Y[tid][k2]; s += y; s2 += y*y; }
    float m = s/HD, var = s2/HD - m*m;
    float rr = rsqrtf(var + 1e-5f);
    rowm[tid] = m; rowr[tid] = rr;
    mrbuf[(size_t)(base+tid)*2+0] = m; mrbuf[(size_t)(base+tid)*2+1] = rr;
  }
  __syncthreads();
#pragma unroll
  for (int i=0;i<8;i++){
    int r = rb+i, n = base+r;
    float xh = (yv[i] - rowm[r]) * rowr[r];
    hnext[(size_t)n*HD + col] = xh*lng[l*HD + col] + lnb[l*HD + col];
  }
}

__global__ void k_temb(const float* __restrict__ t, const float* __restrict__ tpW,
                       const float* __restrict__ tpb, float* __restrict__ temb){
  int i = blockIdx.x*blockDim.x + threadIdx.x;
  if (i >= BATCHB*HD) return;
  int b = i/HD, o = i%HD;
  float acc = tpb[o];
  float tb = t[b];
  const float c0 = -logf(10000.0f)/(HD/2 - 1);
  for (int k=0;k<HD;k++){
    int kk = (k < 64) ? k : k-64;
    float te = tb*__expf(c0*kk);
    float f  = (k < 64) ? __sinf(te) : __cosf(te);
    acc += f*tpW[k*HD + o];
  }
  temb[i] = acc;
}

__global__ void k_add_temb(float* __restrict__ h, const int* __restrict__ batch,
                           const float* __restrict__ temb){
  int i = blockIdx.x*blockDim.x + threadIdx.x;
  if (i >= NATOMS*HD) return;
  int n = i/HD, c = i%HD;
  h[i] = h[i] + temb[batch[n]*HD + c];
}

// Energy head backward: gvec[n][j] = silu'(h@enW1+b1)[j]*enW2[j]; G = gvec@enW1^T
__global__ void k_en_g(const float* __restrict__ hF, const float* __restrict__ enW1,
                       const float* __restrict__ enb1, const float* __restrict__ enW2,
                       float* __restrict__ gvec){
  int i = blockIdx.x*blockDim.x + threadIdx.x;
  if (i >= NATOMS*64) return;
  int n = i/64, j = i%64;
  float z = enb1[j];
  for (int k=0;k<HD;k++) z += hF[(size_t)n*HD + k]*enW1[k*64 + j];
  gvec[i] = dsilu_f(z)*enW2[j];
}
__global__ void k_en_G(const float* __restrict__ gvec, const float* __restrict__ enW1,
                       float* __restrict__ G){
  int i = blockIdx.x*blockDim.x + threadIdx.x;
  if (i >= NATOMS*HD) return;
  int n = i/HD, k = i%HD;
  float a = 0.f;
  for (int j=0;j<64;j++) a += gvec[(size_t)n*64 + j]*enW1[k*64 + j];
  G[i] = a;
}

// Type head.
__global__ void k_ty1(const float* __restrict__ hF, const float* __restrict__ W1,
                      const float* __restrict__ b1, float* __restrict__ s1){
  int i = blockIdx.x*blockDim.x + threadIdx.x;
  if (i >= NATOMS*HD) return;
  int n = i/HD, j = i%HD;
  float z = b1[j];
  for (int k=0;k<HD;k++) z += hF[(size_t)n*HD + k]*W1[k*HD + j];
  s1[i] = silu_f(z);
}
__global__ void k_ty2(const float* __restrict__ s1, const float* __restrict__ W2,
                      const float* __restrict__ b2, float* __restrict__ out){
  int i = blockIdx.x*blockDim.x + threadIdx.x;
  if (i >= NATOMS*NTYPE) return;
  int n = i/NTYPE, o = i%NTYPE;
  float a = b2[o];
  for (int k=0;k<HD;k++) a += s1[(size_t)n*HD + k]*W2[k*NTYPE + o];
  out[i] = a;
}

// f_mag per edge + stress outer-product scatter (one 128-thread block per edge).
__global__ void k_stress(const float* __restrict__ hF, const int* __restrict__ ei,
                         const int* __restrict__ batch, const float* __restrict__ fW1,
                         const float* __restrict__ fb1, const float* __restrict__ fW2,
                         const float* __restrict__ fb2, const float* __restrict__ evec,
                         const float* __restrict__ edist, float* __restrict__ vlat){
  __shared__ float cat[256];
  __shared__ float red[128];
  int e = blockIdx.x, tid = threadIdx.x;
  int s = ei[e], d = ei[NEDGE + e];
  cat[tid]       = hF[(size_t)s*HD + tid];
  cat[128 + tid] = hF[(size_t)d*HD + tid];
  __syncthreads();
  float u = fb1[tid];
  for (int k=0;k<256;k++) u += cat[k]*fW1[k*HD + tid];
  red[tid] = silu_f(u)*fW2[tid];
  __syncthreads();
  for (int off=64; off>0; off>>=1){ if (tid < off) red[tid] += red[tid+off]; __syncthreads(); }
  if (tid < 9){
    float fm = red[0] + fb2[0];
    float dist = edist[e];
    int a = tid/3, b = tid%3;
    float dir = evec[e*3 + b]/(dist + 1e-6f);
    atomicAdd(&vlat[batch[s]*9 + tid], evec[e*3 + a]*fm*dir);
  }
}
__global__ void k_vlat(const float* __restrict__ vlat, float* __restrict__ out){
  int i = blockIdx.x*blockDim.x + threadIdx.x;
  if (i >= BATCHB*9) return;
  int b = i/9, r = (i%9)/3, c = i%3;
  out[i] = 0.5f*(vlat[b*9 + r*3 + c] + vlat[b*9 + c*3 + r]);
}

// Node backward: LN bwd -> nW2^T -> silu' -> nW1^T (WMMA); writes Gprev, dagg.
// Backward contractions use ORIGINAL [K][N] weights, which are [N][K] here.
__global__ void k_node_bwd(int l, const float* __restrict__ G, const float* __restrict__ ybuf,
                           const float* __restrict__ mrbuf, const float* __restrict__ ubuf,
                           const float* __restrict__ lng, const _Float16* __restrict__ w16,
                           float* __restrict__ Gprev, float* __restrict__ dagg){
  __shared__ float    DY[16][HD];
  __shared__ __align__(16) _Float16 DY16[16][HD];
  __shared__ __align__(16) _Float16 DU16[16][HD];
  __shared__ float s1s[16], s2s[16];
  int base = blockIdx.x*16, tid = threadIdx.x;
  if (tid < 16){
    int n = base + tid;
    float m = mrbuf[(size_t)n*2], rr = mrbuf[(size_t)n*2+1];
    float a = 0.f, b = 0.f;
    for (int k=0;k<HD;k++){
      float t  = G[(size_t)n*HD + k]*lng[l*HD + k];
      float xh = (ybuf[(size_t)n*HD + k] - m)*rr;
      a += t; b += t*xh;
    }
    s1s[tid] = a/HD; s2s[tid] = b/HD;
  }
  __syncthreads();
  for (int idx = tid; idx < 16*HD; idx += 256){
    int r = idx >> 7, c = idx & 127;
    int n = base + r;
    float m = mrbuf[(size_t)n*2], rs = mrbuf[(size_t)n*2+1];
    float t  = G[(size_t)n*HD + c]*lng[l*HD + c];
    float xh = (ybuf[(size_t)n*HD + c] - m)*rs;
    float dy = (t - s1s[r] - xh*s2s[r])*rs;
    DY[r][c] = dy; DY16[r][c] = (_Float16)dy;
  }
  __syncthreads();
  int wave = tid >> 5, lane = tid & 31, nb = wave*16;
  const _Float16* W2o = w16 + O_N2 + (size_t)l*HD*HD;      // [i][j]: N=i, K=j
  const _Float16* W1o = w16 + O_N1 + (size_t)l*2*HD*HD;    // [c][j]: N=c, K=j
  v8f c = {};
  for (int k=0;k<HD;k+=32) c = wmma32(ldA(&DY16[0][0], HD, k), ldBc(W2o, HD, k, nb), c);
  int col = nb + (lane & 15), rb = (lane & 16) ? 8 : 0;
#pragma unroll
  for (int i=0;i<8;i++){
    int n = base + rb + i;
    float du = c[i]*dsilu_f(ubuf[(size_t)n*HD + col]);
    DU16[rb+i][col] = (_Float16)du;
  }
  __syncthreads();
  for (int t2=0;t2<2;t2++){
    int nb2 = (wave*2 + t2)*16;                 // 16 column-tiles over 256 outputs
    v8f cc = {};
    for (int k=0;k<HD;k+=32) cc = wmma32(ldA(&DU16[0][0], HD, k), ldBc(W1o, HD, k, nb2), cc);
    int c2 = nb2 + (lane & 15);
#pragma unroll
    for (int i=0;i<8;i++){
      int r = rb+i, n = base+r;
      if (c2 < HD) Gprev[(size_t)n*HD + c2] = DY[r][c2] + cc[i];   // residual + front half
      else         dagg [(size_t)n*HD + (c2-HD)] = cc[i];           // back half
    }
  }
}

// Edge backward: recompute edge MLP (WMMA), scatter dh[src], accumulate dgauss.
// Stages transposed (recompute) + original (backward) weights via async copies.
__global__ void k_edge_bwd(int l, const _Float16* __restrict__ gauss,
                           const _Float16* __restrict__ w16,
                           const float* __restrict__ eb1, const float* __restrict__ eb2,
                           const int* __restrict__ ei, const float* __restrict__ h,
                           const float* __restrict__ dagg, float* __restrict__ Gprev,
                           float* __restrict__ dgacc){
  __shared__ __align__(16) _Float16 Gt[16][NGPAD];
  __shared__ __align__(16) _Float16 S16[16][HD];
  __shared__ float    SP[16][HD];
  __shared__ __align__(16) _Float16 DW16[16][HD];
  __shared__ __align__(16) _Float16 DZ16[16][HD];
  __shared__ __align__(16) _Float16 W1Ts[HD*NGPAD];  // eW1^T [j][g]
  __shared__ __align__(16) _Float16 W2Ts[HD*HD];     // eW2^T [j][i]
  __shared__ __align__(16) _Float16 W1o[NGPAD*HD];   // eW1   [g][j]
  __shared__ __align__(16) _Float16 W2o[HD*HD];      // eW2   [i][j]
  __shared__ int srcS[16], dstS[16];
  int base = blockIdx.x*16, tid = threadIdx.x;
  async_g2l(&Gt[0][0], gauss + (size_t)base*NGPAD, 16*NGPAD);
  async_g2l(W1Ts, w16 + O_E1T + (size_t)l*HD*NGPAD, HD*NGPAD);
  async_g2l(W2Ts, w16 + O_E2T + (size_t)l*HD*HD, HD*HD);
  async_g2l(W1o,  w16 + O_E1  + (size_t)l*NGPAD*HD, NGPAD*HD);
  async_g2l(W2o,  w16 + O_E2  + (size_t)l*HD*HD, HD*HD);
  if (tid < 16){ srcS[tid] = ei[base+tid]; dstS[tid] = ei[NEDGE+base+tid]; }
  if (tid == 0 && base + 16 < NEDGE)
    __builtin_prefetch(gauss + (size_t)(base+16)*NGPAD, 0, 1);
  async_wait_barrier();
  int wave = tid >> 5, lane = tid & 31, nb = wave*16;
  v8f c = {};
  for (int k=0;k<NGPAD;k+=32) c = wmma32(ldA(&Gt[0][0], NGPAD, k), ldBc(W1Ts, NGPAD, k, nb), c);
  int col = nb + (lane & 15), rb = (lane & 16) ? 8 : 0;
#pragma unroll
  for (int i=0;i<8;i++){
    float z = c[i] + eb1[l*HD + col];
    S16[rb+i][col] = (_Float16)silu_f(z);
    SP [rb+i][col] = dsilu_f(z);
  }
  __syncthreads();
  v8f c2 = {};
  for (int k=0;k<HD;k+=32) c2 = wmma32(ldA(&S16[0][0], HD, k), ldBc(W2Ts, HD, k, nb), c2);
#pragma unroll
  for (int i=0;i<8;i++){
    int r = rb+i;
    float we = c2[i] + eb2[l*HD + col];
    float dm = dagg[(size_t)dstS[r]*HD + col];
    float hs = h   [(size_t)srcS[r]*HD + col];
    atomicAdd(&Gprev[(size_t)srcS[r]*HD + col], dm*we);
    DW16[r][col] = (_Float16)(dm*hs);
  }
  __syncthreads();
  v8f c3 = {};
  for (int k=0;k<HD;k+=32) c3 = wmma32(ldA(&DW16[0][0], HD, k), ldBc(W2o, HD, k, nb), c3);
#pragma unroll
  for (int i=0;i<8;i++) DZ16[rb+i][col] = (_Float16)(c3[i]*SP[rb+i][col]);
  __syncthreads();
  if (wave < 4){                               // 64 gaussian outputs = 4 tiles
    int nbg = wave*16;
    v8f c4 = {};
    for (int k=0;k<HD;k+=32) c4 = wmma32(ldA(&DZ16[0][0], HD, k), ldBc(W1o, HD, k, nbg), c4);
    int g = nbg + (lane & 15);
#pragma unroll
    for (int i=0;i<8;i++){
      size_t o = (size_t)(base+rb+i)*NGPAD + g;
      dgacc[o] += c4[i];                       // unique (edge,g) per lane; layers run serially
    }
  }
}

// dfrac from h0 path: M[a][o] = sum_k coordW[a][k]*combW[128+k][o]; dfrac = G0 @ M^T
__global__ void k_M(const float* __restrict__ coordW, const float* __restrict__ combW,
                    float* __restrict__ M){
  int i = blockIdx.x*blockDim.x + threadIdx.x;
  if (i >= 3*HD) return;
  int j = i/HD, o = i%HD;
  float a = 0.f;
  for (int k=0;k<HD;k++) a += coordW[j*HD + k]*combW[(size_t)(HD+k)*HD + o];
  M[i] = a;
}
__global__ void k_dfrac_direct(const float* __restrict__ G, const float* __restrict__ M,
                               float* __restrict__ dfrac){
  int n = blockIdx.x*blockDim.x + threadIdx.x;
  if (n >= NATOMS) return;
#pragma unroll
  for (int j=0;j<3;j++){
    float a = 0.f;
    for (int o=0;o<HD;o++) a += G[(size_t)n*HD + o]*M[j*HD + o];
    dfrac[n*3 + j] = a;
  }
}

// Gaussian/geometry backward: dgauss -> ddist -> dvec -> lattice^T -> dfrac scatter.
__global__ void k_gauss_bwd(const float* __restrict__ dgacc, const float* __restrict__ edist,
                            const float* __restrict__ evec, const float* __restrict__ lattice,
                            const int* __restrict__ ei, const int* __restrict__ batch,
                            float* __restrict__ dfrac){
  int e = blockIdx.x*blockDim.x + threadIdx.x;
  if (e >= NEDGE) return;
  int s = ei[e], d = ei[NEDGE + e];
  int b = batch[s];
  float dist = edist[e];
  const float step  = CUTV/(NGAUSS-1);
  const float coeff = -0.5f/(step*step);
  float dd = 0.f;
  for (int g=0; g<NGAUSS; g++){
    float t  = dist - step*g;
    float gv = __expf(coeff*t*t);
    dd += dgacc[(size_t)e*NGPAD + g]*2.f*coeff*t*gv;
  }
  float inv = dd / fmaxf(dist, 1e-12f);
  float dv[3];
#pragma unroll
  for (int k=0;k<3;k++) dv[k] = evec[e*3 + k]*inv;
  const float* lat = lattice + b*9;
#pragma unroll
  for (int j=0;j<3;j++){
    float dx = dv[0]*lat[j*3+0] + dv[1]*lat[j*3+1] + dv[2]*lat[j*3+2];
    atomicAdd(&dfrac[s*3 + j],  dx);
    atomicAdd(&dfrac[d*3 + j], -dx);
  }
}

__global__ void k_vcoords(const float* __restrict__ dfrac, float* __restrict__ out){
  int i = blockIdx.x*blockDim.x + threadIdx.x;
  if (i >= NATOMS*3) return;
  out[i] = -dfrac[i];
}

// ---------------------------------------------------------------------------
extern "C" void kernel_launch(void* const* d_in, const int* in_sizes, int n_in,
                              void* d_out, int out_size, void* d_ws, size_t ws_size,
                              hipStream_t stream) {
  const int*   x_t     = (const int*)  d_in[0];
  const float* frac    = (const float*)d_in[1];
  const float* lattice = (const float*)d_in[2];
  const int*   ei      = (const int*)  d_in[3];
  const float* tvec    = (const float*)d_in[4];
  const int*   batch   = (const int*)  d_in[5];
  const float* atom_emb= (const float*)d_in[6];
  const float* coordW  = (const float*)d_in[7];
  const float* coordb  = (const float*)d_in[8];
  const float* combW   = (const float*)d_in[9];
  const float* combb   = (const float*)d_in[10];
  const float* eW1     = (const float*)d_in[11];
  const float* eb1     = (const float*)d_in[12];
  const float* eW2     = (const float*)d_in[13];
  const float* eb2     = (const float*)d_in[14];
  const float* nW1     = (const float*)d_in[15];
  const float* nb1     = (const float*)d_in[16];
  const float* nW2     = (const float*)d_in[17];
  const float* nb2     = (const float*)d_in[18];
  const float* ln_g    = (const float*)d_in[19];
  const float* ln_b    = (const float*)d_in[20];
  const float* tpW     = (const float*)d_in[21];
  const float* tpb     = (const float*)d_in[22];
  const float* enW1    = (const float*)d_in[23];
  const float* enb1    = (const float*)d_in[24];
  const float* enW2    = (const float*)d_in[25];
  const float* enb2    = (const float*)d_in[26];  (void)enb2;
  const float* tyW1    = (const float*)d_in[27];
  const float* tyb1    = (const float*)d_in[28];
  const float* tyW2    = (const float*)d_in[29];
  const float* tyb2    = (const float*)d_in[30];
  const float* fW1     = (const float*)d_in[31];
  const float* fb1     = (const float*)d_in[32];
  const float* fW2     = (const float*)d_in[33];
  const float* fb2     = (const float*)d_in[34];
  (void)in_sizes; (void)n_in; (void)out_size; (void)ws_size;

  char* wsb = (char*)d_ws; size_t off = 0;
  auto take = [&](size_t bytes)->char*{
    char* p = wsb + off; off = (off + bytes + 255) & ~(size_t)255; return p;
  };
  _Float16* w16   = (_Float16*)take((size_t)W16_TOT*2);
  _Float16* gauss = (_Float16*)take((size_t)NEDGE*NGPAD*2);
  float* evec  = (float*)take((size_t)NEDGE*3*4);
  float* edist = (float*)take((size_t)NEDGE*4);
  float* hbuf  = (float*)take((size_t)(NL+1)*NATOMS*HD*4);
  float* ubuf  = (float*)take((size_t)NL*NATOMS*HD*4);
  float* ybuf  = (float*)take((size_t)NL*NATOMS*HD*4);
  float* mrbuf = (float*)take((size_t)NL*NATOMS*2*4);
  float* aggbuf= (float*)take((size_t)NATOMS*HD*4);
  float* temb  = (float*)take((size_t)BATCHB*HD*4);
  float* gvec  = (float*)take((size_t)NATOMS*64*4);
  float* Ga    = (float*)take((size_t)NATOMS*HD*4);
  float* Gb    = (float*)take((size_t)NATOMS*HD*4);
  float* daggb = (float*)take((size_t)NATOMS*HD*4);
  float* dgacc = (float*)take((size_t)NEDGE*NGPAD*4);
  float* dfrac = (float*)take((size_t)NATOMS*3*4);
  float* s1b   = (float*)take((size_t)NATOMS*HD*4);
  float* Mb    = (float*)take((size_t)3*HD*4);
  float* vlat  = (float*)take((size_t)BATCHB*9*4);

  float* out    = (float*)d_out;
  float* out_vc = out;                                       // N*3
  float* out_vt = out + (size_t)NATOMS*3;                    // N*NTYPE
  float* out_vl = out + (size_t)NATOMS*3 + (size_t)NATOMS*NTYPE;

  dim3 blk(256);
  // Weight conversion (both orientations) + geometry
  k_prep_w16<<<dim3((W16_TOT+255)/256), blk, 0, stream>>>(combW, eW1, eW2, nW1, nW2, w16);
  k_edge_geom<<<dim3((NEDGE+255)/256), blk, 0, stream>>>(frac, lattice, ei, batch, evec, edist, gauss);
  // Embedding GEMM
  k_h0<<<dim3(NATOMS/16), blk, 0, stream>>>(x_t, frac, atom_emb, coordW, coordb, w16, combb, hbuf);
  // Message-passing layers
  for (int l=0; l<NL; l++){
    float* hl = hbuf + (size_t)l*NATOMS*HD;
    float* hn = hbuf + (size_t)(l+1)*NATOMS*HD;
    k_zero<<<dim3((NATOMS*HD+255)/256), blk, 0, stream>>>(aggbuf, NATOMS*HD);
    k_edge_fwd<<<dim3(NEDGE/16), blk, 0, stream>>>(l, gauss, w16, eb1, eb2, ei, hl, aggbuf);
    k_node_fwd<<<dim3(NATOMS/16), blk, 0, stream>>>(l, hl, aggbuf, w16, nb1, nb2, ln_g, ln_b,
        ubuf + (size_t)l*NATOMS*HD, ybuf + (size_t)l*NATOMS*HD,
        mrbuf + (size_t)l*NATOMS*2, hn);
  }
  float* hF = hbuf + (size_t)NL*NATOMS*HD;
  k_temb<<<dim3((BATCHB*HD+255)/256), blk, 0, stream>>>(tvec, tpW, tpb, temb);
  k_add_temb<<<dim3((NATOMS*HD+255)/256), blk, 0, stream>>>(hF, batch, temb);
  // Heads
  k_en_g<<<dim3((NATOMS*64+255)/256), blk, 0, stream>>>(hF, enW1, enb1, enW2, gvec);
  k_en_G<<<dim3((NATOMS*HD+255)/256), blk, 0, stream>>>(gvec, enW1, Ga);
  k_ty1<<<dim3((NATOMS*HD+255)/256), blk, 0, stream>>>(hF, tyW1, tyb1, s1b);
  k_ty2<<<dim3((NATOMS*NTYPE+255)/256), blk, 0, stream>>>(s1b, tyW2, tyb2, out_vt);
  k_zero<<<dim3(2), blk, 0, stream>>>(vlat, BATCHB*9);
  k_stress<<<dim3(NEDGE), dim3(128), 0, stream>>>(hF, ei, batch, fW1, fb1, fW2, fb2, evec, edist, vlat);
  k_vlat<<<dim3(2), blk, 0, stream>>>(vlat, out_vl);
  // Backward through the layers
  k_zero<<<dim3(((size_t)NEDGE*NGPAD+255)/256), blk, 0, stream>>>(dgacc, NEDGE*NGPAD);
  float* Gcur = Ga; float* Gprev = Gb;
  for (int l=NL-1; l>=0; l--){
    float* hl = hbuf + (size_t)l*NATOMS*HD;
    k_node_bwd<<<dim3(NATOMS/16), blk, 0, stream>>>(l, Gcur,
        ybuf + (size_t)l*NATOMS*HD, mrbuf + (size_t)l*NATOMS*2,
        ubuf + (size_t)l*NATOMS*HD, ln_g, w16, Gprev, daggb);
    k_edge_bwd<<<dim3(NEDGE/16), blk, 0, stream>>>(l, gauss, w16, eb1, eb2, ei,
        hl, daggb, Gprev, dgacc);
    float* tmp = Gcur; Gcur = Gprev; Gprev = tmp;
  }
  // dfrac: direct embedding path + gaussian/geometry path
  k_M<<<dim3(2), blk, 0, stream>>>(coordW, combW, Mb);
  k_dfrac_direct<<<dim3((NATOMS+255)/256), blk, 0, stream>>>(Gcur, Mb, dfrac);
  k_gauss_bwd<<<dim3((NEDGE+255)/256), blk, 0, stream>>>(dgacc, edist, evec, lattice, ei, batch, dfrac);
  k_vcoords<<<dim3((NATOMS*3+255)/256), blk, 0, stream>>>(dfrac, out_vc);
}
